// CrossAttentionFusionHead_30167850287198
// MI455X (gfx1250) — matrix operand order repaired and checked by reading, not compiled, not run on hardware
//
#include <hip/hip_runtime.h>
#include <math.h>

typedef __attribute__((ext_vector_type(16))) _Float16 v16h;
typedef __attribute__((ext_vector_type(8)))  float    v8f;

union Frag16 { v16h v; uint4 u[2]; };

#define ACT_NONE    0
#define ACT_SIGMOID 2

// ---------------------------------------------------------------------------
// Batched GEMM via v_wmma_f32_16x16x32_f16. One wave computes a 16x64 strip
// (4 C tiles, 4 independent accumulators -> pipelined WMMAs).
//   A: f32, k-contiguous:      A(row,k)  = A[z*bsA + row*rsA + k]
//   B: f16, k-contiguous pack: B(k,col)  = Bh[z*bsB + col*ldb + k]
//   C: act(alpha*acc + bias[col]) at C[z*bsC + row*rsC + col*csC] (f32 or f16)
// M % 16 == 0, N % 64 == 0, K % 32 == 0 for every call below.
// ---------------------------------------------------------------------------
template <int ACT, bool F16OUT>
__global__ __launch_bounds__(32) void wmma_gemm_kernel(
    const float* __restrict__ A, const _Float16* __restrict__ Bh,
    const float* __restrict__ bias, void* __restrict__ Cv, int K,
    long long rsA, long long bsA,
    long long ldb, long long bsB,
    long long rsC, long long csC, long long bsC,
    float alpha)
{
    const int tn = blockIdx.x;   // 64-wide output col block
    const int tm = blockIdx.y;   // 16-row output tile
    const int z  = blockIdx.z;   // batch

    const int lane = threadIdx.x & 31;
    const int hl   = lane & 15;
    const int hi   = lane >> 4;

    const float*    pArow = A  + (long long)z * bsA + (long long)(tm * 16 + hl) * rsA;
    const _Float16* Bb    = Bh + (long long)z * bsB;

    v8f acc[4];
    #pragma unroll
    for (int ct = 0; ct < 4; ++ct) acc[ct] = (v8f){0.f,0.f,0.f,0.f,0.f,0.f,0.f,0.f};

    for (int kb = 0; kb < K; kb += 32) {
        // A 16x32 f16 fragment: lane(hi) holds K in {8*hi+0..7} U {16+8*hi+0..7}
        const float4 f0 = *(const float4*)(pArow + kb + 8 * hi);
        const float4 f1 = *(const float4*)(pArow + kb + 8 * hi + 4);
        const float4 f2 = *(const float4*)(pArow + kb + 16 + 8 * hi);
        const float4 f3 = *(const float4*)(pArow + kb + 16 + 8 * hi + 4);
        v16h a;
        a[0]  = (_Float16)f0.x; a[1]  = (_Float16)f0.y;
        a[2]  = (_Float16)f0.z; a[3]  = (_Float16)f0.w;
        a[4]  = (_Float16)f1.x; a[5]  = (_Float16)f1.y;
        a[6]  = (_Float16)f1.z; a[7]  = (_Float16)f1.w;
        a[8]  = (_Float16)f2.x; a[9]  = (_Float16)f2.y;
        a[10] = (_Float16)f2.z; a[11] = (_Float16)f2.w;
        a[12] = (_Float16)f3.x; a[13] = (_Float16)f3.y;
        a[14] = (_Float16)f3.z; a[15] = (_Float16)f3.w;

        #pragma unroll
        for (int ct = 0; ct < 4; ++ct) {
            // B 32x16 f16 fragment: lane holds col=hl, K in {8*hi+0..7} U {16+8*hi+0..7}
            const _Float16* bp =
                Bb + (long long)(tn * 64 + ct * 16 + hl) * ldb + kb;
            Frag16 b;
            b.u[0] = *(const uint4*)(bp + 8 * hi);
            b.u[1] = *(const uint4*)(bp + 16 + 8 * hi);
            acc[ct] = __builtin_amdgcn_wmma_f32_16x16x32_f16(
                false, a, false, b.v, (short)0, acc[ct], false, false);
        }
    }

    float*    C32 = (float*)Cv    + (long long)z * bsC;
    _Float16* C16 = (_Float16*)Cv + (long long)z * bsC;
    #pragma unroll
    for (int ct = 0; ct < 4; ++ct) {
        const int col = tn * 64 + ct * 16 + hl;
        const float bv = bias ? bias[col] : 0.f;
        #pragma unroll
        for (int v = 0; v < 8; ++v) {
            float x = alpha * acc[ct][v] + bv;
            if (ACT == ACT_SIGMOID) x = __builtin_amdgcn_rcpf(1.f + __expf(-x));
            // C/D layout: VGPR v -> row v + 8*hi, col = hl
            const long long off = (long long)(tm * 16 + v + 8 * hi) * rsC
                                + (long long)col * csC;
            if (F16OUT) C16[off] = (_Float16)x;
            else        C32[off] = x;
        }
    }
}

// dimg = img - tgt (elementwise)
__global__ void sub_kernel(const float* __restrict__ a, const float* __restrict__ b,
                           float* __restrict__ c, int n)
{
    int i = blockIdx.x * blockDim.x + threadIdx.x;
    if (i < n) c[i] = a[i] - b[i];
}

// dst[c][r] = f16(src[r][c])  : pack a weight to f16, k-contiguous per column
__global__ void packT_kernel(const float* __restrict__ src, _Float16* __restrict__ dst,
                             int R, int C)
{
    int idx = blockIdx.x * blockDim.x + threadIdx.x;
    if (idx < R * C) {
        int c = idx / R, r = idx - c * R;
        dst[idx] = (_Float16)src[r * C + c];
    }
}

// bo1[j] = b1[j] + sum_e bo[e] * W1[e,j]   (768 -> 256)
__global__ void bo1_kernel(const float* __restrict__ bo, const float* __restrict__ W1,
                           const float* __restrict__ b1, float* __restrict__ bo1)
{
    int j = threadIdx.x;  // 256 threads
    float s = b1[j];
    for (int e = 0; e < 768; ++e) s += bo[e] * W1[e * 256 + j];
    bo1[j] = s;
}

// ---------------------------------------------------------------------------
// Fused tail. Block = 256 threads (8 waves) handles 16 n x 8 m = 128 pairs.
// Phase 1: h1[p][j] = relu(Vt1[m] + sum_h w[n,m,h]*Dv1[m,h,j]) -> LDS (f16)
// Phase 2: (128x256) @ W2t(256x64) with WMMA, ReLU(+b2)
// Phase 3: dot with W3, half-wave shfl reduce, tanh*0.5 -> out[n][m]
// ---------------------------------------------------------------------------
__global__ __launch_bounds__(256) void fused_tail_kernel(
    const float* __restrict__ Vt1, const float* __restrict__ Dv1,
    const float* __restrict__ wgt, const _Float16* __restrict__ W2t,
    const float* __restrict__ b2, const float* __restrict__ W3,
    const float* __restrict__ b3, float* __restrict__ out)
{
    __shared__ __align__(16) _Float16 h1s[128 * 256];  // 64 KB
    const int n0 = blockIdx.x * 16;
    const int m0 = blockIdx.y * 8;
    const int t  = threadIdx.x;

    {   // ---- phase 1: rank-8 combine (2 threads per pair, 128 j each) ----
        const int p  = t >> 1;
        const int j0 = (t & 1) * 128;
        const int n  = n0 + (p & 15);
        const int m  = m0 + (p >> 4);
        float wv[8];
        const float* wp = wgt + (long long)n * 4096 + (long long)m * 8;
        #pragma unroll
        for (int h = 0; h < 8; ++h) wv[h] = wp[h];
        const float* vt  = Vt1 + (long long)m * 256;
        const float* dvp = Dv1 + (long long)m * 2048;
        for (int j = j0; j < j0 + 128; j += 4) {
            float4 x = *(const float4*)(vt + j);
            #pragma unroll
            for (int h = 0; h < 8; ++h) {
                const float4 d = *(const float4*)(dvp + h * 256 + j);
                x.x += wv[h] * d.x; x.y += wv[h] * d.y;
                x.z += wv[h] * d.z; x.w += wv[h] * d.w;
            }
            h1s[p * 256 + j + 0] = (_Float16)fmaxf(x.x, 0.f);
            h1s[p * 256 + j + 1] = (_Float16)fmaxf(x.y, 0.f);
            h1s[p * 256 + j + 2] = (_Float16)fmaxf(x.z, 0.f);
            h1s[p * 256 + j + 3] = (_Float16)fmaxf(x.w, 0.f);
        }
    }
    __syncthreads();

    // ---- phase 2+3: wave g owns pairs [16g, 16g+16) ----
    const int lane = t & 31, g = t >> 5;
    const int hl = lane & 15, hi = lane >> 4;
    float sv[8] = {0.f, 0.f, 0.f, 0.f, 0.f, 0.f, 0.f, 0.f};
    const _Float16* ap = &h1s[(g * 16 + hl) * 256];

    #pragma unroll
    for (int ct = 0; ct < 4; ++ct) {
        const int col = ct * 16 + hl;
        v8f acc = {0.f, 0.f, 0.f, 0.f, 0.f, 0.f, 0.f, 0.f};
        const _Float16* bp = W2t + col * 256;
        for (int kb = 0; kb < 256; kb += 32) {
            Frag16 a, b;
            a.u[0] = *(const uint4*)(ap + kb + 8 * hi);
            a.u[1] = *(const uint4*)(ap + kb + 16 + 8 * hi);
            b.u[0] = *(const uint4*)(bp + kb + 8 * hi);
            b.u[1] = *(const uint4*)(bp + kb + 16 + 8 * hi);
            acc = __builtin_amdgcn_wmma_f32_16x16x32_f16(false, a.v, false, b.v,
                                                         (short)0, acc, false, false);
        }
        const float w3c = W3[col], b2c = b2[col];
        #pragma unroll
        for (int v = 0; v < 8; ++v)
            sv[v] += fmaxf(acc[v] + b2c, 0.f) * w3c;  // relu(h1@W2+b2) . W3
    }
    #pragma unroll
    for (int off = 1; off < 16; off <<= 1) {
        #pragma unroll
        for (int v = 0; v < 8; ++v) sv[v] += __shfl_xor(sv[v], off, 32);
    }
    if (hl == 0) {
        const float bb = b3[0];
        #pragma unroll
        for (int v = 0; v < 8; ++v) {
            const int p = g * 16 + v + 8 * hi;
            const int n = n0 + (p & 15), m = m0 + (p >> 4);
            out[(long long)n * 512 + m] = tanhf(sv[v] + bb) * 0.5f;
        }
    }
}

extern "C" void kernel_launch(void* const* d_in, const int* in_sizes, int n_in,
                              void* d_out, int out_size, void* d_ws, size_t ws_size,
                              hipStream_t stream)
{
    (void)in_sizes; (void)n_in; (void)out_size; (void)ws_size;
    const float* qe  = (const float*)d_in[0];
    const float* ie  = (const float*)d_in[1];
    const float* te  = (const float*)d_in[2];
    const float* Wq  = (const float*)d_in[3];
    const float* bq  = (const float*)d_in[4];
    const float* Wi  = (const float*)d_in[5];
    const float* bi  = (const float*)d_in[6];
    const float* Wt  = (const float*)d_in[7];
    const float* bt  = (const float*)d_in[8];
    const float* Wiq = (const float*)d_in[9];
    const float* biq = (const float*)d_in[10];
    const float* Wik = (const float*)d_in[11];
    const float* Wiv = (const float*)d_in[13];
    const float* biv = (const float*)d_in[14];
    const float* Wo  = (const float*)d_in[15];
    const float* bo  = (const float*)d_in[16];
    const float* W1  = (const float*)d_in[17];
    const float* b1  = (const float*)d_in[18];
    const float* W2  = (const float*)d_in[19];
    const float* b2  = (const float*)d_in[20];
    const float* W3  = (const float*)d_in[21];
    const float* b3  = (const float*)d_in[22];
    float* out = (float*)d_out;

    // ---- workspace layout: f32 region then f16 region (~26 MB total) ----
    float* ws   = (float*)d_ws;
    float* q    = ws;              // 256*768
    float* img  = q    + 196608;   // 512*768
    float* tgt  = img  + 393216;   // 512*768
    float* qh   = tgt  + 393216;   // 256*768 (unscaled; scale folded into logits)
    float* dimg = qh   + 196608;   // 512*768
    float* dv   = dimg + 393216;   // 512*768
    float* vtg  = dv   + 393216;   // 512*768
    float* bo1v = vtg  + 393216;   // 256      (bo @ W1 + b1)
    float* Vt1  = bo1v + 256;      // 512*256
    float* Dv1  = Vt1  + 131072;   // 512*8*256
    float* wv   = Dv1  + 1048576;  // 256*512*8 sigmoid weights
    _Float16* h16  = (_Float16*)(wv + 1048576);
    _Float16* Wqt  = h16;              // [768][768]
    _Float16* Wit  = Wqt  + 589824;
    _Float16* Wtt  = Wit  + 589824;
    _Float16* Wiqt = Wtt  + 589824;
    _Float16* Wikt = Wiqt + 589824;
    _Float16* Wivt = Wikt + 589824;
    _Float16* W1t  = Wivt + 589824;    // [256][768]
    _Float16* W2t  = W1t  + 196608;    // [64][256]
    _Float16* Wo1t = W2t  + 16384;     // [256][768]  (Wo@W1)^T, f16
    _Float16* dkh  = Wo1t + 196608;    // [512][768]  (img-tgt)@Wik, f16

    const dim3 wblk(32);
    const dim3 eblk(256);
    const float one = 1.0f;
    const float lscale = 0.1020620726159658f;  // 1/sqrt(96)

    // ---- pack weights to f16, k-contiguous per output column ----
    packT_kernel<<<dim3(2304), eblk, 0, stream>>>(Wq,  Wqt,  768, 768);
    packT_kernel<<<dim3(2304), eblk, 0, stream>>>(Wi,  Wit,  768, 768);
    packT_kernel<<<dim3(2304), eblk, 0, stream>>>(Wt,  Wtt,  768, 768);
    packT_kernel<<<dim3(2304), eblk, 0, stream>>>(Wiq, Wiqt, 768, 768);
    packT_kernel<<<dim3(2304), eblk, 0, stream>>>(Wik, Wikt, 768, 768);
    packT_kernel<<<dim3(2304), eblk, 0, stream>>>(Wiv, Wivt, 768, 768);
    packT_kernel<<<dim3(768),  eblk, 0, stream>>>(W1,  W1t,  768, 256);
    packT_kernel<<<dim3(64),   eblk, 0, stream>>>(W2,  W2t,  256, 64);

    // ---- input projections ----
    wmma_gemm_kernel<ACT_NONE,false><<<dim3(12,16,1), wblk, 0, stream>>>(
        qe, Wqt, bq, q, 768,  768,0, 768,0, 768,1,0, one);
    wmma_gemm_kernel<ACT_NONE,false><<<dim3(12,32,1), wblk, 0, stream>>>(
        ie, Wit, bi, img, 768,  768,0, 768,0, 768,1,0, one);
    wmma_gemm_kernel<ACT_NONE,false><<<dim3(12,32,1), wblk, 0, stream>>>(
        te, Wtt, bt, tgt, 768,  768,0, 768,0, 768,1,0, one);
    // MHA q projection (unscaled; scale folded into logits)
    wmma_gemm_kernel<ACT_NONE,false><<<dim3(12,16,1), wblk, 0, stream>>>(
        q, Wiqt, biq, qh, 768,  768,0, 768,0, 768,1,0, one);
    // dimg = img - tgt;  dk = dimg@Wik (f16 out), dv = dimg@Wiv  (biases cancel)
    sub_kernel<<<dim3(1536), eblk, 0, stream>>>(img, tgt, dimg, 393216);
    wmma_gemm_kernel<ACT_NONE,true ><<<dim3(12,32,1), wblk, 0, stream>>>(
        dimg, Wikt, nullptr, dkh, 768,  768,0, 768,0, 768,1,0, one);
    wmma_gemm_kernel<ACT_NONE,false><<<dim3(12,32,1), wblk, 0, stream>>>(
        dimg, Wivt, nullptr, dv, 768,  768,0, 768,0, 768,1,0, one);
    wmma_gemm_kernel<ACT_NONE,false><<<dim3(12,32,1), wblk, 0, stream>>>(
        tgt, Wivt, biv, vtg, 768,  768,0, 768,0, 768,1,0, one);
    // Wo1t = (Wo@W1)^T in f16: C stored transposed via rsC=1, csC=768
    wmma_gemm_kernel<ACT_NONE,true ><<<dim3(4,48,1), wblk, 0, stream>>>(
        Wo, W1t, nullptr, Wo1t, 768,  768,0, 768,0, 1,768,0, one);
    bo1_kernel<<<dim3(1), eblk, 0, stream>>>(bo, W1, b1, bo1v);
    // Vt1 = v_tgt @ Wo1 + bo1
    wmma_gemm_kernel<ACT_NONE,false><<<dim3(4,32,1), wblk, 0, stream>>>(
        vtg, Wo1t, bo1v, Vt1, 768,  768,0, 768,0, 256,1,0, one);
    // Dv1[m,h,:] = dv[m, 96h:96h+96] @ Wo1[96h:96h+96, :]   (batched over h)
    wmma_gemm_kernel<ACT_NONE,false><<<dim3(4,32,8), wblk, 0, stream>>>(
        dv, Wo1t, nullptr, Dv1, 96,  768,96, 768,96, 2048,1,256, one);
    // w[n,m,h] = sigmoid(scale * qh_h[n] . dk_h[m])   (batched over h)
    wmma_gemm_kernel<ACT_SIGMOID,false><<<dim3(8,16,8), wblk, 0, stream>>>(
        qh, dkh, nullptr, wv, 96,  768,96, 768,96, 4096,8,1, lscale);
    // fused tail: combine + MLP + tanh
    fused_tail_kernel<<<dim3(16,64,1), eblk, 0, stream>>>(
        Vt1, Dv1, wv, W2t, b2, W3, b3, out);
}
